// LinkModel_66571993088847
// MI455X (gfx1250) — matrix operand order repaired and checked by reading, not compiled, not run on hardware
//
#include <hip/hip_runtime.h>
#include <hip/hip_bf16.h>
#include <math.h>

typedef __attribute__((ext_vector_type(16))) _Float16 v16h;
typedef __attribute__((ext_vector_type(8)))  _Float16 v8h;
typedef __attribute__((ext_vector_type(8)))  float    v8f;

#define HDIM 512
#define LP_K 1024
#define LP_PAD 8   // pad row stride by 8 halfs (16B) -> conflict-free LDS banks

// ---------------------------------------------------------------------------
// WMMA helpers (CDNA5 wave32, v_wmma_f32_16x16x32_f16, f32 accumulate)
// ---------------------------------------------------------------------------
__device__ __forceinline__ v8f wmma_f16(v16h a, v16h b, v8f c) {
  // 8 args: (neg_a, A, neg_b, B, c_mod, C, reuse_a, reuse_b)
  return __builtin_amdgcn_wmma_f32_16x16x32_f16(false, a, false, b, (short)0, c,
                                                false, false);
}

// A-fragment (16x32 f16, row-major source; works for global or LDS rows).
// Lane L holds row M=L&15. Lanes 0-15: K = kb+{0..7},{16..23}; lanes 16-31: +8.
__device__ __forceinline__ v16h load_frag_a(const _Float16* row, int kb, int hi) {
  const v8h a0 = *(const v8h*)(row + kb + hi * 8);
  const v8h a1 = *(const v8h*)(row + kb + hi * 8 + 16);
  return __builtin_shufflevector(a0, a1, 0, 1, 2, 3, 4, 5, 6, 7,
                                 8, 9, 10, 11, 12, 13, 14, 15);
}

// B-fragment from pre-transposed Bt [N, K] f16. Lane L holds col N=n0+(L&15),
// K = kb+{0..7},{16..23} (lanes 0-15) or +8 (lanes 16-31): two 16B loads.
__device__ __forceinline__ v16h load_frag_b(const _Float16* __restrict__ Bt,
                                            int K, int n0, int kb, int lane) {
  const int n = lane & 15, hi = lane >> 4;
  return load_frag_a(Bt + (size_t)(n0 + n) * K, kb, hi);
}

// ---------------------------------------------------------------------------
// Small prep kernels
// ---------------------------------------------------------------------------
__global__ void k_fill1(float* __restrict__ deg, int n) {
  int i = blockIdx.x * blockDim.x + threadIdx.x;
  if (i < n) deg[i] = 1.0f;  // self loop
}

__global__ void k_deg_edges(const int* __restrict__ dst, float* __restrict__ deg,
                            int E) {
  int i = blockIdx.x * blockDim.x + threadIdx.x;
  if (i < E) atomicAdd(&deg[dst[i]], 1.0f);
}

__global__ void k_dis(float* __restrict__ deg, int n) {
  int i = blockIdx.x * blockDim.x + threadIdx.x;
  if (i < n) deg[i] = rsqrtf(deg[i]);  // deg >= 1 always (self loops)
}

__global__ void k_norm(const int* __restrict__ src, const int* __restrict__ dst,
                       const float* __restrict__ dis, float* __restrict__ norm,
                       int E) {
  int i = blockIdx.x * blockDim.x + threadIdx.x;
  if (i < E) norm[i] = dis[src[i]] * dis[dst[i]];
}

__global__ void k_cvt_f16(const float* __restrict__ in, _Float16* __restrict__ out,
                          size_t n) {
  size_t i = (size_t)blockIdx.x * blockDim.x + threadIdx.x;
  if (i < n) out[i] = (_Float16)in[i];
}

__global__ void k_relu_cvt_f16(const float* __restrict__ in,
                               _Float16* __restrict__ out, size_t n) {
  size_t i = (size_t)blockIdx.x * blockDim.x + threadIdx.x;
  if (i < n) {
    float v = in[i];
    out[i] = (_Float16)(v > 0.0f ? v : 0.0f);
  }
}

// W[K,N] f32 row-major -> Wt[N,K] f16
__global__ void k_transpose_cvt(const float* __restrict__ W,
                                _Float16* __restrict__ Wt, int K, int N) {
  int i = blockIdx.x * blockDim.x + threadIdx.x;
  if (i < K * N) {
    int k = i / N, n = i % N;
    Wt[(size_t)n * K + k] = (_Float16)W[i];
  }
}

// agg = bias + xw * dis^2   (self-loop contribution, race-free full write)
__global__ void k_bias_self(const float* __restrict__ xw,
                            const float* __restrict__ bias,
                            const float* __restrict__ dis,
                            float* __restrict__ agg, size_t total) {
  size_t i = (size_t)blockIdx.x * blockDim.x + threadIdx.x;
  if (i < total) {
    int node = (int)(i >> 9);      // /512
    int c = (int)(i & 511);
    float d = dis[node];
    agg[i] = bias[c] + xw[i] * d * d;
  }
}

// edge scatter: agg[dst] += xw[src] * norm, 4 floats per thread
__global__ void k_scatter(const int* __restrict__ src, const int* __restrict__ dst,
                          const float* __restrict__ norm,
                          const float* __restrict__ xw, float* __restrict__ agg,
                          int E) {
  size_t idx = (size_t)blockIdx.x * blockDim.x + threadIdx.x;
  size_t total = (size_t)E * (HDIM / 4);
  if (idx >= total) return;
  int e = (int)(idx >> 7);           // / (512/4)
  int c = (int)((idx & 127) << 2);
  float nm = norm[e];
  const float4 v = *(const float4*)(xw + (size_t)src[e] * HDIM + c);
  float* o = agg + (size_t)dst[e] * HDIM + c;
  atomicAdd(o + 0, v.x * nm);
  atomicAdd(o + 1, v.y * nm);
  atomicAdd(o + 2, v.z * nm);
  atomicAdd(o + 3, v.w * nm);
}

// ---------------------------------------------------------------------------
// WMMA GEMM: C[M,N] f32 = A[M,K] f16 @ Bt[N,K] f16
// block = 8 waves; block b computes rows [16b,16b+16); wave w -> cols [64w,64w+64)
// Per K-step: batch A + all 4 B fragment loads first, then 4 WMMAs, so the
// compiler can issue one load clause and overlap waits with matrix ops.
// Branch-free A prefetch 256B ahead (speculative; OOB translations dropped).
// ---------------------------------------------------------------------------
__global__ void __launch_bounds__(256)
k_gemm(const _Float16* __restrict__ A, const _Float16* __restrict__ Bt,
       float* __restrict__ C, int M, int N, int K) {
  const int lane = threadIdx.x & 31;
  const int w = threadIdx.x >> 5;
  const int m0 = blockIdx.x * 16;
  const int n0 = w * 64;
  if (m0 >= M || n0 >= N) return;  // wave-uniform
  const int m = lane & 15, hi = lane >> 4;
  const _Float16* arow = A + (size_t)(m0 + m) * K;

  v8f acc[4] = {};
  for (int kb = 0; kb < K; kb += 32) {
    __builtin_prefetch(arow + kb + 128, 0, 1);  // global_prefetch_b8, no branch
    v16h a = load_frag_a(arow, kb, hi);
    v16h bf[4];
#pragma unroll
    for (int j = 0; j < 4; ++j)
      bf[j] = load_frag_b(Bt, K, n0 + j * 16, kb, lane);
#pragma unroll
    for (int j = 0; j < 4; ++j)
      acc[j] = wmma_f16(a, bf[j], acc[j]);
  }
#pragma unroll
  for (int j = 0; j < 4; ++j) {
    float* cp = C + (size_t)(m0 + hi * 8) * N + n0 + j * 16 + m;
#pragma unroll
    for (int e = 0; e < 8; ++e) cp[(size_t)e * N] = acc[j][e];
  }
}

// ---------------------------------------------------------------------------
// Fused link predictor: 16 pairs per block, 8 waves.
// Gathered A panel concat(z[sp], z[tp]) [16 x 1024] f16 is staged once per
// block into LDS with GLOBAL_LOAD_ASYNC_TO_LDS_B128 (async engine, no VGPR
// round-trip, tracked by ASYNCcnt), then the hot loop reads A fragments via
// ds_load_b128 (DScnt) while B fragments stream from global (LOADcnt) --
// independent counters overlap naturally. Epilogue: relu(h1+b1) dot lpW2
// in-register, cross-lane shfl_xor reduce, LDS accumulate, sigmoid.
// ---------------------------------------------------------------------------
__global__ void __launch_bounds__(256)
k_lp(const _Float16* __restrict__ Z, const _Float16* __restrict__ W1t,  // [512,1024]
     const float* __restrict__ b1, const float* __restrict__ w2,        // [512]
     const float* __restrict__ b2, const int* __restrict__ sp,
     const int* __restrict__ tp, float* __restrict__ out, int P) {
  __shared__ _Float16 As[16][LP_K + LP_PAD];   // ~33 KB, padded row stride
  __shared__ float logits[16];
  const int t = threadIdx.x;
  const int lane = t & 31;
  const int w = t >> 5;
  const int p0 = blockIdx.x * 16;
  const int m = lane & 15, hi = lane >> 4;

  if (t < 16) logits[t] = b2[0];

  // Stage 16 gathered rows (2 KB each = 128 x 16B chunks) cooperatively via
  // async global->LDS DMA. chunk < 64 -> first half from z[sp[row]], else
  // second half from z[tp[row]].
#pragma unroll
  for (int i = 0; i < 8; ++i) {
    int idx = i * 256 + t;          // 0..2047
    int r = idx >> 7;               // row 0..15
    int ch = idx & 127;             // 16B chunk 0..127
    int node = (ch < 64) ? sp[p0 + r] : tp[p0 + r];
    const _Float16* gp = Z + (size_t)node * HDIM + (size_t)(ch & 63) * 8;
    unsigned int laddr = (unsigned int)(uintptr_t)(&As[r][ch * 8]);
    asm volatile("global_load_async_to_lds_b128 %0, %1, off"
                 :: "v"(laddr), "v"(gp)
                 : "memory");
  }
  asm volatile("s_wait_asynccnt 0x0" ::: "memory");
  __syncthreads();

  const int n0 = w * 64;
  v8f acc[4] = {};
  for (int kb = 0; kb < LP_K; kb += 32) {   // kb uniform -> EXEC stays full
    v16h a = load_frag_a(&As[m][0], kb, hi);   // ds_load_b128 x2
    v16h bf[4];
#pragma unroll
    for (int j = 0; j < 4; ++j)
      bf[j] = load_frag_b(W1t, LP_K, n0 + j * 16, kb, lane);
#pragma unroll
    for (int j = 0; j < 4; ++j)
      acc[j] = wmma_f16(a, bf[j], acc[j]);
  }

  // rows handled by this lane: M = hi*8 + e ; col = n0 + j*16 + m
  float psum[8];
#pragma unroll
  for (int e = 0; e < 8; ++e) psum[e] = 0.0f;
#pragma unroll
  for (int j = 0; j < 4; ++j) {
    int col = n0 + j * 16 + m;
    float bb = b1[col], ww = w2[col];
#pragma unroll
    for (int e = 0; e < 8; ++e) {
      float v = acc[j][e] + bb;
      v = v > 0.0f ? v : 0.0f;
      psum[e] += v * ww;
    }
  }
  // reduce across the 16 lanes of each half (xor masks 1..8 stay within half)
#pragma unroll
  for (int e = 0; e < 8; ++e) {
    float v = psum[e];
    for (int mask = 1; mask <= 8; mask <<= 1) v += __shfl_xor(v, mask, 32);
    psum[e] = v;
  }
  if (m == 0) {  // lanes 0 and 16 of each wave
#pragma unroll
    for (int e = 0; e < 8; ++e) atomicAdd(&logits[hi * 8 + e], psum[e]);
  }
  __syncthreads();
  if (t < 16 && p0 + t < P) {
    float x = logits[t];
    out[p0 + t] = 1.0f / (1.0f + __expf(-x));
  }
}

// ---------------------------------------------------------------------------
// Host launch
// ---------------------------------------------------------------------------
extern "C" void kernel_launch(void* const* d_in, const int* in_sizes, int n_in,
                              void* d_out, int out_size, void* d_ws, size_t ws_size,
                              hipStream_t stream) {
  const float* x    = (const float*)d_in[0];
  const int*   ei   = (const int*)d_in[1];
  const int*   ep   = (const int*)d_in[2];
  const float* W1   = (const float*)d_in[3];
  const float* b1   = (const float*)d_in[4];
  const float* W2   = (const float*)d_in[5];
  const float* b2   = (const float*)d_in[6];
  const float* lpW1 = (const float*)d_in[7];
  const float* lpb1 = (const float*)d_in[8];
  const float* lpW2 = (const float*)d_in[9];
  const float* lpb2 = (const float*)d_in[10];
  float* out = (float*)d_out;

  const int H = HDIM;
  const int n = in_sizes[0] / H;
  const int E = in_sizes[1] / 2;
  const int P = in_sizes[2] / 2;
  const int* esrc = ei;
  const int* edst = ei + E;
  const int* psrc = ep;
  const int* ptgt = ep + P;

  // workspace carving
  char* ws = (char*)d_ws;
  size_t off = 0;
  auto carve = [&](size_t bytes) -> char* {
    char* p = ws + off;
    off = (off + bytes + 255) & ~(size_t)255;
    return p;
  };
  float*    dis    = (float*)carve((size_t)n * 4);
  float*    enorm  = (float*)carve((size_t)E * 4);
  _Float16* Wt1    = (_Float16*)carve((size_t)H * H * 2);
  _Float16* Wt2    = (_Float16*)carve((size_t)H * H * 2);
  _Float16* lpW1t  = (_Float16*)carve((size_t)2 * H * H * 2);
  _Float16* A16    = (_Float16*)carve((size_t)n * H * 2);   // f16 GEMM input / z
  float*    xw     = (float*)carve((size_t)n * H * 4);      // GEMM output
  float*    agg    = (float*)carve((size_t)n * H * 4);      // aggregation

  const int T = 256;
  const size_t nh = (size_t)n * H;
  auto blk = [](size_t work, int t) { return (int)((work + t - 1) / t); };

  // 1) normalization coefficients
  k_fill1<<<blk(n, T), T, 0, stream>>>(dis, n);
  k_deg_edges<<<blk(E, T), T, 0, stream>>>(edst, dis, E);
  k_dis<<<blk(n, T), T, 0, stream>>>(dis, n);
  k_norm<<<blk(E, T), T, 0, stream>>>(esrc, edst, dis, enorm, E);

  // 2) operand prep
  k_cvt_f16<<<blk(nh, T), T, 0, stream>>>(x, A16, nh);
  k_transpose_cvt<<<blk((size_t)H * H, T), T, 0, stream>>>(W1, Wt1, H, H);
  k_transpose_cvt<<<blk((size_t)H * H, T), T, 0, stream>>>(W2, Wt2, H, H);
  k_transpose_cvt<<<blk((size_t)2 * H * H, T), T, 0, stream>>>(lpW1, lpW1t, 2 * H, H);

  const int gemm_blocks = (n + 15) / 16;
  const size_t scat = (size_t)E * (H / 4);

  // 3) layer 1: xw = x@W1 ; agg = b1 + self + edge-scatter ; relu -> f16
  k_gemm<<<gemm_blocks, 256, 0, stream>>>(A16, Wt1, xw, n, H, H);
  k_bias_self<<<blk(nh, T), T, 0, stream>>>(xw, b1, dis, agg, nh);
  k_scatter<<<blk(scat, T), T, 0, stream>>>(esrc, edst, enorm, xw, agg, E);
  k_relu_cvt_f16<<<blk(nh, T), T, 0, stream>>>(agg, A16, nh);

  // 4) layer 2: xw = relu(z1)@W2 ; agg = b2 + self + edge-scatter ; -> f16 z
  k_gemm<<<gemm_blocks, 256, 0, stream>>>(A16, Wt2, xw, n, H, H);
  k_bias_self<<<blk(nh, T), T, 0, stream>>>(xw, b2, dis, agg, nh);
  k_scatter<<<blk(scat, T), T, 0, stream>>>(esrc, edst, enorm, xw, agg, E);
  k_cvt_f16<<<blk(nh, T), T, 0, stream>>>(agg, A16, nh);  // z (no relu)

  // 5) fused link predictor
  k_lp<<<(P + 15) / 16, 256, 0, stream>>>(A16, lpW1t, lpb1, lpW2, lpb2,
                                          psrc, ptgt, out, P);
}
